// MambaDriverDistraction_3435973836967
// MI455X (gfx1250) — matrix operand order repaired and checked by reading, compiled, not run on hardware
//
#include <hip/hip_runtime.h>

// ---------------- model constants ----------------
#define NBLK   6
#define BATCH  16
#define IMG    384
#define PATCH  16
#define PGRID  24            // IMG/PATCH
#define NPATCH (PGRID*PGRID) // 576
#define NTOK   (NPATCH + 1)  // 577
#define DM     256           // D_MODEL
#define DI     512           // D_INNER
#define DSTATE 64
#define DTRANK 16
#define DCONV  4
#define LALL   (BATCH*NTOK)  // 9232 rows (multiple of 16)
#define MPATCH (BATCH*NPATCH)// 9216
#define KPATCH (3*PATCH*PATCH) // 768
#define NCLS   10
#define SCHUNK 8             // tokens staged per TDM transfer in the scan

typedef __attribute__((ext_vector_type(16))) __bf16    v16bf;
typedef __attribute__((ext_vector_type(8)))  float     v8f;
typedef __attribute__((ext_vector_type(4)))  unsigned  uv4;
typedef __attribute__((ext_vector_type(4)))  int       iv4;
typedef __attribute__((ext_vector_type(8)))  int       iv8;

__device__ __forceinline__ unsigned short f2bf(float f) {
  unsigned u = __float_as_uint(f);
  return (unsigned short)((u + 0x7fffu + ((u >> 16) & 1u)) >> 16); // RNE truncate
}
__device__ __forceinline__ float sigmoidf_(float x) { return 1.0f / (1.0f + __expf(-x)); }

// ---------------- TDM: DMA a [rows x 128 f32] tile (row stride 144 f32) into LDS ----------------
// Builds the D# descriptor per CDNA5 ISA 8.3/8.4: count=1, type=2 ("image"),
// data_size=4B, tensor_dim0=128 (stride 144), tile = 128 x SCHUNK, 2D (groups 2/3 zero).
// tensor_dim1 = rows_avail so out-of-bounds rows zero-fill.
__device__ __forceinline__ void tdm_load_bc(const float* gsrc, float* ldsdst, int rows_avail) {
  unsigned long long ga = (unsigned long long)gsrc;
  unsigned lds = (unsigned)(unsigned long long)ldsdst;   // LDS byte offset = addr[31:0]
  unsigned td1 = (unsigned)rows_avail;
  uv4 g0;
  g0.x = 1u;                                             // count=1, user mode
  g0.y = lds;                                            // lds_addr
  g0.z = (unsigned)(ga & 0xFFFFFFFFull);                 // global_addr[31:0]
  g0.w = (unsigned)((ga >> 32) & 0x1FFFFFFull) | (2u << 30); // global_addr[56:32] | type=2
  iv8 g1;
  g1[0] = 0x00020000;                                    // data_size=2 (4 bytes)
  g1[1] = (int)(128u << 16);                             // tensor_dim0[15:0] -> bits 63:48
  g1[2] = (int)((td1 & 0xFFFFu) << 16);                  // tensor_dim0[31:16]=0 | tensor_dim1[15:0]
  g1[3] = (int)((td1 >> 16) | (128u << 16));             // tensor_dim1[31:16] | tile_dim0=128
  g1[4] = (int)SCHUNK;                                   // tile_dim1=SCHUNK, tile_dim2=0
  g1[5] = 144;                                           // tensor_dim0_stride[31:0]
  g1[6] = 0;                                             // stride0[47:32] | stride1[15:0]
  g1[7] = 0;
  iv4 z4 = {0,0,0,0};
#if defined(__clang_major__) && (__clang_major__ >= 23)
  iv8 z8 = {0,0,0,0,0,0,0,0};
  __builtin_amdgcn_tensor_load_to_lds(g0, g1, z4, z4, z8, 0);
#else
  __builtin_amdgcn_tensor_load_to_lds(g0, g1, z4, z4, 0);
#endif
}

// ---------------- weight convert: [N,K] f32 -> [Kpad,Npad] bf16 (transposed, zero-padded) ----
__global__ void k_wT(const float* __restrict__ src, unsigned short* __restrict__ dst,
                     int N, int K, int Npad, int Kpad) {
  size_t t = (size_t)blockIdx.x * blockDim.x + threadIdx.x;
  size_t total = (size_t)Kpad * Npad;
  if (t >= total) return;
  int k = (int)(t / Npad), n = (int)(t % Npad);
  const float* s = src + (size_t)blockIdx.y * N * K;
  unsigned short* d = dst + (size_t)blockIdx.y * total;
  float v = (k < K && n < N) ? s[(size_t)n*K + k] : 0.0f;
  d[t] = f2bf(v);
}

// ---------------- im2col for patch embedding ----------------
__global__ void k_im2col(const float* __restrict__ x, unsigned short* __restrict__ patches) {
  size_t t = (size_t)blockIdx.x * blockDim.x + threadIdx.x;
  const size_t total = (size_t)MPATCH * KPATCH;
  if (t >= total) return;
  int col = (int)(t % KPATCH);
  size_t row = t / KPATCH;
  int b  = (int)(row / NPATCH);
  int p  = (int)(row % NPATCH);
  int py = p / PGRID, px = p % PGRID;
  int c  = col / (PATCH*PATCH);
  int r  = col % (PATCH*PATCH);
  int iy = r / PATCH, ix = r % PATCH;
  float v = x[(((size_t)b*3 + c)*IMG + (py*PATCH + iy))*IMG + (px*PATCH + ix)];
  patches[t] = f2bf(v);
}

// ---------------- token assembly: cls + pos_embed ----------------
__global__ void k_assemble(const float* __restrict__ tmp, const float* __restrict__ cls,
                           const float* __restrict__ pos, float* __restrict__ tok) {
  size_t t = (size_t)blockIdx.x * blockDim.x + threadIdx.x;
  const size_t total = (size_t)LALL * DM;
  if (t >= total) return;
  int d = (int)(t % DM);
  size_t row = t / DM;
  int b = (int)(row / NTOK);
  int l = (int)(row % NTOK);
  float v = (l == 0) ? cls[d] : tmp[((size_t)b*NPATCH + (l-1))*DM + d];
  tok[t] = v + pos[(size_t)l*DM + d];
}

// ---------------- LayerNorm (one wave per 256-wide row) -> bf16 ----------------
__global__ __launch_bounds__(32) void k_layernorm_bf16(const float* __restrict__ x,
                                                       const float* __restrict__ w,
                                                       const float* __restrict__ b,
                                                       unsigned short* __restrict__ out,
                                                       int row_mult) {
  int lane = threadIdx.x;
  size_t row = (size_t)blockIdx.x * row_mult;
  const float* xr = x + row * DM;
  float v[8]; float s = 0.f;
  #pragma unroll
  for (int j = 0; j < 8; ++j) { v[j] = xr[j*32 + lane]; s += v[j]; }
  #pragma unroll
  for (int o = 16; o > 0; o >>= 1) s += __shfl_xor(s, o, 32);
  float mean = s * (1.0f/DM);
  float vs = 0.f;
  #pragma unroll
  for (int j = 0; j < 8; ++j) { float d = v[j]-mean; vs += d*d; }
  #pragma unroll
  for (int o = 16; o > 0; o >>= 1) vs += __shfl_xor(vs, o, 32);
  float rstd = rsqrtf(vs * (1.0f/DM) + 1e-5f);
  #pragma unroll
  for (int j = 0; j < 8; ++j) {
    int cidx = j*32 + lane;
    float o = (v[j]-mean)*rstd*w[cidx] + b[cidx];
    out[(size_t)blockIdx.x*DM + cidx] = f2bf(o);
  }
}

// ---------------- WMMA bf16 GEMM: C[M,N] = act(A[M,Kpad] * WT[Kpad,Npad]) ----------------
// One wave computes a 16 x (16*NTW) C tile: A fragment reused across NTW WMMAs.
// A loads are contiguous 16B (global_load_b128), B is gathered per 16-bit element; K fully unrolled.
// MODE: 0 = none, 1 = +bias, 2 = softplus(acc+bias), 3 = +resid
template<int KP, int NP, int NTW, int MODE, bool NGUARD>
__global__ __launch_bounds__(32) void k_gemm(const unsigned short* __restrict__ A,
                                             const unsigned short* __restrict__ WT,
                                             float* __restrict__ C, int ldc,
                                             const float* __restrict__ bias,
                                             const float* __restrict__ resid,
                                             int Nreal) {
  const int m0 = blockIdx.x * 16;
  const int n0 = blockIdx.y * 16 * NTW;
  const int lane = threadIdx.x;
  const int mrow = m0 + (lane & 15);
  const int half = lane >> 4;

  v8f acc[NTW];
  #pragma unroll
  for (int t = 0; t < NTW; ++t) acc[t] = (v8f){0.f,0.f,0.f,0.f,0.f,0.f,0.f,0.f};
  union OPU { uint4 q[2]; v16bf v; };

  const unsigned short* arow = A + (size_t)mrow * KP + half*8;

  #pragma unroll
  for (int k0 = 0; k0 < KP; k0 += 32) {
    // A operand: ISA 7.12.2 16-bit A 16x32 layout -> two contiguous 16B chunks per lane
    OPU a;
    a.q[0] = *(const uint4*)(arow + k0);
    a.q[1] = *(const uint4*)(arow + k0 + 16);
    // B operands: lane holds column n0+t*16+(lane&15), K rows k0+half*8+{0..7} then k0+16+half*8+{0..7}
    const unsigned short* brow = WT + (size_t)(k0 + half*8) * NP + n0 + (lane & 15);
    OPU b[NTW];
    #pragma unroll
    for (int t = 0; t < NTW; ++t) {
      #pragma unroll
      for (int s = 0; s < 16; ++s)
        ((unsigned short*)&b[t])[s] = brow[(size_t)((s & 7) + (s >> 3) * 16) * NP + t*16];
    }
    if (k0 + 32 < KP)
      __builtin_prefetch(WT + (size_t)(k0 + 32 + lane) * NP + n0, 0, 1); // global_prefetch
    #pragma unroll
    for (int t = 0; t < NTW; ++t)
      acc[t] = __builtin_amdgcn_wmma_f32_16x16x32_bf16(false, a.v, false, b[t].v,
                                                       (short)0, acc[t], false, false);
  }

  // C/D layout: VGPR r holds M = r + 8*half, N = n0 + t*16 + (lane&15)
  #pragma unroll
  for (int t = 0; t < NTW; ++t) {
    const int n = n0 + t*16 + (lane & 15);
    if (!NGUARD || n < Nreal) {
      #pragma unroll
      for (int r = 0; r < 8; ++r) {
        int m = m0 + r + 8*half;
        float v = acc[t][r];
        if (MODE == 1)      v += bias[n];
        else if (MODE == 2) { v += bias[n]; v = (v > 20.f) ? v : __logf(1.0f + __expf(v)); }
        else if (MODE == 3) v += resid[(size_t)m*ldc + n];
        C[(size_t)m*ldc + n] = v;
      }
    }
  }
}

// ---------------- causal depthwise conv (K=4) + SiLU ----------------
__global__ void k_conv_silu(const float* __restrict__ xz, const float* __restrict__ cw,
                            const float* __restrict__ cb, float* __restrict__ xi,
                            unsigned short* __restrict__ xibf) {
  size_t t = (size_t)blockIdx.x * blockDim.x + threadIdx.x;
  const size_t total = (size_t)LALL * DI;
  if (t >= total) return;
  int c = (int)(t & (DI-1));
  size_t m = t >> 9;             // b*577 + l
  int l = (int)(m % NTOK);
  float acc = cb[c];
  #pragma unroll
  for (int k = 0; k < DCONV; ++k) {
    int ls = l - (DCONV-1) + k;
    if (ls >= 0) acc += cw[c*DCONV + k] * xz[(m - (DCONV-1) + k) * 1024 + c];
  }
  float s = acc * sigmoidf_(acc);
  xi[t]   = s;
  xibf[t] = f2bf(s);
}

// ---------------- dtr slice of dbl -> bf16 padded to 32 cols ----------------
__global__ void k_dtr_bf16(const float* __restrict__ dbl, unsigned short* __restrict__ dtrbf) {
  size_t t = (size_t)blockIdx.x * blockDim.x + threadIdx.x;
  const size_t total = (size_t)LALL * 32;
  if (t >= total) return;
  size_t m = t >> 5; int k = (int)(t & 31);
  dtrbf[t] = (k < DTRANK) ? f2bf(dbl[m*144 + k]) : (unsigned short)0;
}

// ---------------- selective scan: TDM double-buffered B_t/C_t staging in LDS ----------------
// One block = 256 channels of one batch (grid = BATCH*2). Wave 0 drives the Tensor
// Data Mover: each transfer stages SCHUNK tokens' [B_t|C_t] rows (128 f32, stride 144)
// into LDS; compute of chunk c overlaps the DMA of chunk c+1 (TENSORcnt + barrier sync).
__global__ __launch_bounds__(256) void k_scan(const float* __restrict__ xi,
                                              const float* __restrict__ dt,
                                              const float* __restrict__ dbl,
                                              const float* __restrict__ A_log,
                                              const float* __restrict__ Dp,
                                              float* __restrict__ y) {
  __shared__ float sStage[2][SCHUNK * 128];
  const int tid = threadIdx.x;
  const int b = blockIdx.x >> 1;
  const int d = ((blockIdx.x & 1) << 8) + tid;   // channel
  float a[DSTATE], h[DSTATE];
  #pragma unroll
  for (int s = 0; s < DSTATE; ++s) { a[s] = -__expf(A_log[(size_t)d*DSTATE + s]); h[s] = 0.f; }
  const float Dd = Dp[d];
  const size_t rowbase = (size_t)b * NTOK;
  const int nch = (NTOK + SCHUNK - 1) / SCHUNK;

  if (tid < 32)   // wave 0 issues the first DMA
    tdm_load_bc(dbl + (rowbase)*144 + DTRANK, &sStage[0][0], NTOK);

  for (int c = 0; c < nch; ++c) {
    if (tid < 32) __builtin_amdgcn_s_wait_tensorcnt(0);   // chunk c landed in LDS
    __syncthreads();
    if (tid < 32 && c + 1 < nch) {                        // prefetch chunk c+1
      int l1 = (c + 1) * SCHUNK;
      tdm_load_bc(dbl + (rowbase + l1)*144 + DTRANK, &sStage[(c + 1) & 1][0], NTOK - l1);
    }
    const float* buf = sStage[c & 1];
    const int l0 = c * SCHUNK;
    const int lim = (NTOK - l0 < SCHUNK) ? (NTOK - l0) : SCHUNK;
    for (int r = 0; r < lim; ++r) {
      size_t row = rowbase + l0 + r;
      float dtv = dt[row*DI + d];
      float uv  = xi[row*DI + d];
      const float* sB = buf + r*128;
      const float* sC = sB + DSTATE;
      float du = dtv * uv;
      float accy = 0.f;
      #pragma unroll
      for (int s = 0; s < DSTATE; ++s) {
        float dA = __expf(dtv * a[s]);
        h[s] = dA * h[s] + du * sB[s];
        accy += h[s] * sC[s];
      }
      y[row*DI + d] = accy + uv * Dd;
    }
    __syncthreads();   // all reads of this buffer done before it is overwritten
  }
}

// ---------------- gate: yg = bf16( y * silu(z) ) ----------------
__global__ void k_gate(const float* __restrict__ y, const float* __restrict__ xz,
                       unsigned short* __restrict__ ygbf) {
  size_t t = (size_t)blockIdx.x * blockDim.x + threadIdx.x;
  const size_t total = (size_t)LALL * DI;
  if (t >= total) return;
  size_t m = t >> 9; int c = (int)(t & (DI-1));
  float z = xz[m*1024 + DI + c];
  ygbf[t] = f2bf(y[t] * z * sigmoidf_(z));
}

// ======================================================================
extern "C" void kernel_launch(void* const* d_in, const int* in_sizes, int n_in,
                              void* d_out, int out_size, void* d_ws, size_t ws_size,
                              hipStream_t stream) {
  (void)in_sizes; (void)n_in; (void)out_size; (void)ws_size;
  const float* x        = (const float*)d_in[0];
  const float* patch_w  = (const float*)d_in[1];
  const float* patch_b  = (const float*)d_in[2];
  const float* cls_tok  = (const float*)d_in[3];
  const float* pos_emb  = (const float*)d_in[4];
  const float* norm_w   = (const float*)d_in[5];
  const float* norm_b   = (const float*)d_in[6];
  const float* in_proj  = (const float*)d_in[7];
  const float* conv_w   = (const float*)d_in[8];
  const float* conv_b   = (const float*)d_in[9];
  const float* x_proj   = (const float*)d_in[10];
  const float* dt_proj  = (const float*)d_in[11];
  const float* dt_pb    = (const float*)d_in[12];
  const float* A_log    = (const float*)d_in[13];
  const float* D_ssm    = (const float*)d_in[14];
  const float* out_proj = (const float*)d_in[15];
  const float* fnorm_w  = (const float*)d_in[16];
  const float* fnorm_b  = (const float*)d_in[17];
  const float* head_w   = (const float*)d_in[18];
  const float* head_b   = (const float*)d_in[19];
  float* out = (float*)d_out;

  // ---- workspace carve-out (256B aligned) ----
  char* ws = (char*)d_ws;
  size_t off = 0;
  auto carve = [&](size_t elems, size_t esz) -> void* {
    off = (off + 255) & ~(size_t)255;
    void* p = ws + off;
    off += elems * esz;
    return p;
  };
  float* tok   = (float*)carve((size_t)LALL*DM, 4);
  float* tmpp  = (float*)carve((size_t)MPATCH*DM, 4);
  float* xz    = (float*)carve((size_t)LALL*1024, 4);
  float* xi    = (float*)carve((size_t)LALL*DI, 4);
  float* dbl   = (float*)carve((size_t)LALL*144, 4);
  float* dtbuf = (float*)carve((size_t)LALL*DI, 4);
  float* ybuf  = (float*)carve((size_t)LALL*DI, 4);
  unsigned short* patches = (unsigned short*)carve((size_t)MPATCH*KPATCH, 2);
  unsigned short* xnorm   = (unsigned short*)carve((size_t)LALL*DM, 2);
  unsigned short* xibf    = (unsigned short*)carve((size_t)LALL*DI, 2);
  unsigned short* dtrbf   = (unsigned short*)carve((size_t)LALL*32, 2);
  unsigned short* ygbf    = (unsigned short*)carve((size_t)LALL*DI, 2);
  unsigned short* clsbf   = (unsigned short*)carve((size_t)16*DM, 2);
  // transposed/padded bf16 weights [Kpad, Npad]
  unsigned short* pwT  = (unsigned short*)carve((size_t)KPATCH*DM, 2);        // 768 x 256
  unsigned short* inwT = (unsigned short*)carve((size_t)NBLK*DM*1024, 2);     // 256 x 1024
  unsigned short* xpwT = (unsigned short*)carve((size_t)NBLK*DI*144, 2);      // 512 x 144
  unsigned short* dtwT = (unsigned short*)carve((size_t)NBLK*32*DI, 2);       // 32  x 512
  unsigned short* owT  = (unsigned short*)carve((size_t)NBLK*DI*DM, 2);       // 512 x 256
  unsigned short* hwT  = (unsigned short*)carve((size_t)DM*16, 2);            // 256 x 16

  const int TB = 256;
  auto cdiv = [](size_t a, size_t b) { return (unsigned)((a + b - 1) / b); };

  // ---- weight transpose + pad + bf16 ----
  { dim3 g(cdiv((size_t)KPATCH*DM,TB), 1);  k_wT<<<g,TB,0,stream>>>(patch_w,  pwT,  DM,  KPATCH, DM,  KPATCH); }
  { dim3 g(cdiv((size_t)DM*1024,TB), NBLK); k_wT<<<g,TB,0,stream>>>(in_proj,  inwT, 1024, DM,   1024, DM); }
  { dim3 g(cdiv((size_t)DI*144,TB),  NBLK); k_wT<<<g,TB,0,stream>>>(x_proj,   xpwT, 144,  DI,   144,  DI); }
  { dim3 g(cdiv((size_t)32*DI,TB),   NBLK); k_wT<<<g,TB,0,stream>>>(dt_proj,  dtwT, DI,   DTRANK, DI, 32); }
  { dim3 g(cdiv((size_t)DI*DM,TB),   NBLK); k_wT<<<g,TB,0,stream>>>(out_proj, owT,  DM,   DI,   DM,  DI); }
  { dim3 g(cdiv((size_t)DM*16,TB), 1);      k_wT<<<g,TB,0,stream>>>(head_w,   hwT,  NCLS, DM,   16,  DM); }

  // ---- patch embedding ----
  k_im2col<<<cdiv((size_t)MPATCH*KPATCH,TB),TB,0,stream>>>(x, patches);
  { dim3 g(MPATCH/16, DM/64);
    k_gemm<KPATCH, DM, 4, 1, false><<<g,32,0,stream>>>(patches, pwT, tmpp, DM,
                                                       patch_b, nullptr, DM); }
  k_assemble<<<cdiv((size_t)LALL*DM,TB),TB,0,stream>>>(tmpp, cls_tok, pos_emb, tok);

  // ---- mamba blocks ----
  for (int blk = 0; blk < NBLK; ++blk) {
    k_layernorm_bf16<<<LALL,32,0,stream>>>(tok, norm_w + blk*DM, norm_b + blk*DM, xnorm, 1);

    { dim3 g(LALL/16, 1024/64);     // in_proj, 16x64 wave tiles
      k_gemm<DM, 1024, 4, 0, false><<<g,32,0,stream>>>(xnorm, inwT + (size_t)blk*DM*1024,
                                                       xz, 1024, nullptr, nullptr, 1024); }

    k_conv_silu<<<cdiv((size_t)LALL*DI,TB),TB,0,stream>>>(xz, conv_w + (size_t)blk*DI*DCONV,
                                                          conv_b + blk*DI, xi, xibf);

    { dim3 g(LALL/16, 144/16);      // x_proj (N=144 -> 16-wide tiles)
      k_gemm<DI, 144, 1, 0, false><<<g,32,0,stream>>>(xibf, xpwT + (size_t)blk*DI*144,
                                                      dbl, 144, nullptr, nullptr, 144); }

    k_dtr_bf16<<<cdiv((size_t)LALL*32,TB),TB,0,stream>>>(dbl, dtrbf);
    { dim3 g(LALL/16, DI/64);       // dt_proj + softplus
      k_gemm<32, DI, 4, 2, false><<<g,32,0,stream>>>(dtrbf, dtwT + (size_t)blk*32*DI,
                                                     dtbuf, DI, dt_pb + blk*DI, nullptr, DI); }

    k_scan<<<BATCH*2,256,0,stream>>>(xi, dtbuf, dbl,
                                     A_log + (size_t)blk*DI*DSTATE, D_ssm + blk*DI, ybuf);

    k_gate<<<cdiv((size_t)LALL*DI,TB),TB,0,stream>>>(ybuf, xz, ygbf);

    { dim3 g(LALL/16, DM/64);       // out_proj + residual
      k_gemm<DI, DM, 4, 3, false><<<g,32,0,stream>>>(ygbf, owT + (size_t)blk*DI*DM,
                                                     tok, DM, nullptr, tok, DM); }
  }

  // ---- final LN on cls rows + head ----
  k_layernorm_bf16<<<BATCH,32,0,stream>>>(tok, fnorm_w, fnorm_b, clsbf, NTOK);
  { dim3 g(BATCH/16, 1);
    k_gemm<DM, 16, 1, 1, true><<<g,32,0,stream>>>(clsbf, hwT, out, NCLS,
                                                  head_b, nullptr, NCLS); }
}
